// MambaMixer_70196945486152
// MI455X (gfx1250) — matrix operand order repaired and checked by reading, not compiled, not run on hardware
//
#include <hip/hip_runtime.h>
#include <hip/hip_bf16.h>
#include <cstdint>

// ---------------- problem constants (match reference) ----------------
#define Bc   2
#define Lc   1024
#define Hc   768
#define DIc  1536
#define Nst  16      // SSM state size N
#define KC   4       // conv kernel
#define Rc   48      // dt rank
#define BLc  (Bc * Lc)     // 2048 rows for all GEMMs
#define KPAD 64            // dt GEMM K padded 48 -> 64

typedef __bf16 bf16_t;
typedef __attribute__((ext_vector_type(16))) __bf16 v16bf;
typedef __attribute__((ext_vector_type(8)))  float  v8f;
typedef __attribute__((ext_vector_type(4)))  unsigned int u32x4;
typedef __attribute__((ext_vector_type(8)))  unsigned int u32x8;

// ---------------- tiled bf16 WMMA GEMM ----------------
// C[m,n] = act( sum_k A[m,k] * W[n,k] + bias[n] )
// A: [M,K] bf16 row-major, W: [N,K] bf16 row-major (i.e. C = A @ W^T)
// Data path: A tile via per-lane global_load_async_to_lds_b128 (ASYNCcnt),
//            W tile via Tensor Data Mover descriptor (TENSORcnt, wave 0),
//            double-buffered LDS, WMMA consumes while next tile is in flight.
#define BM 128
#define BN 128
#define BK 32

#define ACT_NONE     0
#define ACT_SILU     1
#define ACT_SOFTPLUS 2

__device__ __forceinline__ float apply_act(float v, int ACT) {
  if (ACT == ACT_SILU)     return v * (1.0f / (1.0f + __expf(-v)));
  if (ACT == ACT_SOFTPLUS) return (v > 20.0f) ? v : log1pf(__expf(v));
  return v;
}

template <int ACT>
__global__ __launch_bounds__(256)
void gemm_bf16_wmma(const bf16_t* __restrict__ A,
                    const bf16_t* __restrict__ W,
                    const float*  __restrict__ bias,
                    float*        __restrict__ outF,
                    bf16_t*       __restrict__ outB,
                    int M, int N, int K) {
  __shared__ bf16_t As[2][BM * BK];
  __shared__ bf16_t Bs[2][BN * BK];

  const int tid  = threadIdx.x;
  const int lane = tid & 31;
  const int w    = tid >> 5;          // 8 waves
  const int wm   = w & 1;             // 2 waves along M
  const int wn   = w >> 1;            // 4 waves along N
  const int mo   = wm * 64;           // wave sub-tile origin in block tile
  const int no   = wn * 32;
  const int l16  = lane & 15;
  const int half = lane >> 4;

  const int m0 = blockIdx.y * BM;
  const int n0 = blockIdx.x * BN;

  // LDS byte offsets (generic shared addr low 32 bits == LDS offset)
  const unsigned asOff = (unsigned)(uintptr_t)(&As[0][0]);
  const unsigned bsOff = (unsigned)(uintptr_t)(&Bs[0][0]);

  // rows of W available for this block tile (TDM zero-fills OOB reads)
  unsigned remN = (unsigned)(N - n0);
  if (remN > BN) remN = BN;

  // ---- async/TDM tile issue: A per-lane async b128, W via one TDM D# ----
  auto issue_tile = [&](int k0, int buf) {
    // A tile: 128 rows x 32 bf16 = 512 x 16B chunks; 2 chunks per thread
#pragma unroll
    for (int it = 0; it < 2; ++it) {
      int idx = tid + it * 256;
      int row = idx >> 2;
      int c   = idx & 3;
      unsigned lds = asOff + (unsigned)((buf * BM * BK + row * BK + c * 8) * 2);
      unsigned long long ga =
          (unsigned long long)(uintptr_t)(A + (size_t)(m0 + row) * K + k0 + c * 8);
      asm volatile("global_load_async_to_lds_b128 %0, %1, off"
                   :: "v"(lds), "v"(ga) : "memory");
    }
    // W tile: Tensor Data Mover, issued once per workgroup (wave 0; wave-uniform)
    if (w == 0) {
      unsigned long long gaddr =
          (unsigned long long)(uintptr_t)(W + (size_t)n0 * K + k0);
      unsigned ldsB = bsOff + (unsigned)(buf * BN * BK * 2);
      union { u32x4 v; unsigned u[4]; } g0;
      union { u32x8 v; unsigned u[8]; } g1;
      // group 0: count=1 | lds_addr | global_addr[56:0] | type=2
      g0.u[0] = 1u;
      g0.u[1] = ldsB;
      g0.u[2] = (unsigned)gaddr;
      g0.u[3] = ((unsigned)(gaddr >> 32) & 0x01FFFFFFu) | (2u << 30);
      // group 1: data_size=2B; tensor_dim0=K, tensor_dim1=remN;
      //          tile 32 x 128; row stride = K elements
      g1.u[0] = (1u << 16);                                   // data_size=1 (2B)
      g1.u[1] = ((unsigned)K & 0xFFFFu) << 16;                // tensor_dim0 lo
      g1.u[2] = (((unsigned)K >> 16) & 0xFFFFu) | (remN << 16); // dim0 hi | dim1 lo
      g1.u[3] = ((unsigned)BK << 16);                         // dim1 hi=0 | tile_dim0=32
      g1.u[4] = (unsigned)BN;                                 // tile_dim1=128 | tile_dim2=0
      g1.u[5] = (unsigned)K;                                  // tensor_dim0_stride lo
      g1.u[6] = 0u;                                           // stride hi | dim1_stride lo
      g1.u[7] = 0u;                                           // dim1_stride hi
      asm volatile("tensor_load_to_lds %0, %1, null, null"
                   :: "s"(g0.v), "s"(g1.v) : "memory");
    }
  };

  v8f acc[4][2];
#pragma unroll
  for (int i = 0; i < 4; ++i)
#pragma unroll
    for (int j = 0; j < 2; ++j)
      acc[i][j] = (v8f){0.f, 0.f, 0.f, 0.f, 0.f, 0.f, 0.f, 0.f};

  const int nTiles = K / BK;   // all K here are multiples of 32
  issue_tile(0, 0);

  for (int t = 0; t < nTiles; ++t) {
    const int buf = t & 1;
    // our own async/TDM writes for tile t must land before the barrier
    asm volatile("s_wait_asynccnt 0x0" ::: "memory");
    __builtin_amdgcn_s_wait_tensorcnt(0);
    __syncthreads();              // tile t visible to all; tile t-1 reads done
    if (t + 1 < nTiles)
      issue_tile((t + 1) * BK, buf ^ 1);   // overlap next tile with compute

    const bf16_t* Ab = &As[buf][0];
    const bf16_t* Bb = &Bs[buf][0];

    // ---- A fragments (ISA 16-bit A 16x32 layout: K-pairs per VGPR,
    //      lane-half selects K octet) ----
    union AFrag { v16bf v; unsigned int u[8]; };
    union BFrag { v16bf v; uint4 q[2]; };
    AFrag af[4];
#pragma unroll
    for (int mi = 0; mi < 4; ++mi) {
      int row = mo + mi * 16 + l16;
      const unsigned short* r = (const unsigned short*)(Ab + row * BK);
#pragma unroll
      for (int vv = 0; vv < 8; ++vv) {
        int kbase = ((vv < 4) ? (2 * vv) : (16 + 2 * (vv - 4))) + half * 8;
        af[mi].u[vv] = *(const unsigned int*)(r + kbase);
      }
    }
    // ---- B fragments: lane = N col, lane-half selects K octet (k 8h..8h+7, 16+8h..16+8h+7) ----
    BFrag bfm[2];
#pragma unroll
    for (int ni = 0; ni < 2; ++ni) {
      int n = no + ni * 16 + l16;
      const uint4* p = (const uint4*)(Bb + n * BK + half * 8);
      bfm[ni].q[0] = p[0];
      bfm[ni].q[1] = p[2];
    }

    // ---- 8 WMMA per wave per k-step ----
#pragma unroll
    for (int mi = 0; mi < 4; ++mi)
#pragma unroll
      for (int ni = 0; ni < 2; ++ni)
        acc[mi][ni] = __builtin_amdgcn_wmma_f32_16x16x32_bf16(
            false, af[mi].v, false, bfm[ni].v,
            (short)0, acc[mi][ni], false, false);
  }

  // ---- epilogue: bias + activation, fp32 and/or bf16 stores ----
#pragma unroll
  for (int mi = 0; mi < 4; ++mi) {
#pragma unroll
    for (int ni = 0; ni < 2; ++ni) {
      int gn = n0 + no + ni * 16 + l16;
      if (gn >= N) continue;
      float bv = (bias != nullptr) ? bias[gn] : 0.0f;
#pragma unroll
      for (int j = 0; j < 8; ++j) {
        int gm = m0 + mo + mi * 16 + half * 8 + j;
        float v = acc[mi][ni][j] + bv;
        v = apply_act(v, ACT);
        size_t o = (size_t)gm * N + gn;
        if (outF) outF[o] = v;
        if (outB) outB[o] = (bf16_t)v;
      }
    }
  }
}

// ---------------- elementwise helpers ----------------
__global__ void f32_to_bf16_kernel(const float* __restrict__ in,
                                   bf16_t* __restrict__ out, int n) {
  int i = blockIdx.x * blockDim.x + threadIdx.x;
  if (i < n) out[i] = (bf16_t)in[i];
}

// depthwise causal conv (K=4) + SiLU.  proj layout: [B, L, 2*DI], hs = cols [0,DI)
__global__ void conv_silu_kernel(const float* __restrict__ proj,
                                 const float* __restrict__ cw,
                                 const float* __restrict__ cb,
                                 float* __restrict__ hscF,
                                 bf16_t* __restrict__ hscB) {
  int i = blockIdx.x * blockDim.x + threadIdx.x;
  if (i >= Bc * Lc * DIc) return;
  int d = i % DIc;
  int l = (i / DIc) % Lc;
  int b = i / (DIc * Lc);
  float acc = cb[d];
#pragma unroll
  for (int j = 0; j < KC; ++j) {
    int ls = l - (KC - 1) + j;
    if (ls >= 0)
      acc += cw[d * KC + j] * proj[((size_t)(b * Lc + ls)) * (2 * DIc) + d];
  }
  float v = acc * (1.0f / (1.0f + __expf(-acc)));
  size_t o = ((size_t)(b * Lc + l)) * DIc + d;
  hscF[o] = v;
  hscB[o] = (bf16_t)v;
}

// ssm_p[:, :R] -> [BL, KPAD] bf16, zero padded
__global__ void pad_ts_kernel(const float* __restrict__ ssmp,
                              bf16_t* __restrict__ out) {
  int i = blockIdx.x * blockDim.x + threadIdx.x;
  if (i >= BLc * KPAD) return;
  int col = i % KPAD, row = i / KPAD;
  out[i] = (col < Rc) ? (bf16_t)ssmp[(size_t)row * (Rc + 2 * Nst) + col]
                      : (bf16_t)0.0f;
}

// dt_w [DI, R] -> [DI, KPAD] bf16, zero padded
__global__ void pad_dtw_kernel(const float* __restrict__ dtw,
                               bf16_t* __restrict__ out) {
  int i = blockIdx.x * blockDim.x + threadIdx.x;
  if (i >= DIc * KPAD) return;
  int col = i % KPAD, row = i / KPAD;
  out[i] = (col < Rc) ? (bf16_t)dtw[(size_t)row * Rc + col] : (bf16_t)0.0f;
}

// ---------------- selective scan ----------------
// grid: (DI/256, B), block 256.  Each thread owns one channel, 16 states in regs.
__global__ __launch_bounds__(256)
void scan_kernel(const float* __restrict__ ssmp,   // [BL, R+2N]
                 const float* __restrict__ dtf,    // [BL, DI] softplus(dt)
                 const float* __restrict__ hscF,   // [BL, DI]
                 const float* __restrict__ projF,  // [BL, 2*DI] (gate in cols DI..)
                 const float* __restrict__ A_log,  // [DI, N]
                 const float* __restrict__ Dp,     // [DI]
                 bf16_t* __restrict__ yB) {        // [BL, DI]
  const int b = blockIdx.y;
  const int d = blockIdx.x * 256 + threadIdx.x;
  __shared__ float Bsh[Nst];
  __shared__ float Csh[Nst];

  float Ad[Nst];
#pragma unroll
  for (int n = 0; n < Nst; ++n) Ad[n] = -__expf(A_log[(size_t)d * Nst + n]);
  const float Dd = Dp[d];
  float s[Nst];
#pragma unroll
  for (int n = 0; n < Nst; ++n) s[n] = 0.0f;

  for (int l = 0; l < Lc; ++l) {
    __syncthreads();                          // WAR vs previous step
    if (threadIdx.x < 2 * Nst) {
      float v = ssmp[((size_t)(b * Lc + l)) * (Rc + 2 * Nst) + Rc + threadIdx.x];
      if (threadIdx.x < Nst) Bsh[threadIdx.x] = v;
      else                   Csh[threadIdx.x - Nst] = v;
    }
    __syncthreads();

    const size_t row = (size_t)(b * Lc + l);
    const float dtv = dtf[row * DIc + d];
    const float u   = hscF[row * DIc + d];
    const float g   = projF[row * (2 * DIc) + DIc + d];
    float y = 0.0f;
#pragma unroll
    for (int n = 0; n < Nst; ++n) {
      float dA = __expf(dtv * Ad[n]);
      s[n] = fmaf(dA, s[n], dtv * Bsh[n] * u);
      y = fmaf(s[n], Csh[n], y);
    }
    float yo = y + u * Dd;
    float sg = g * (1.0f / (1.0f + __expf(-g)));
    yB[row * DIc + d] = (bf16_t)(yo * sg);
  }
}

// ---------------- host launch ----------------
static inline int ceil_div(int a, int b) { return (a + b - 1) / b; }

extern "C" void kernel_launch(void* const* d_in, const int* in_sizes, int n_in,
                              void* d_out, int out_size, void* d_ws, size_t ws_size,
                              hipStream_t stream) {
  const float* x        = (const float*)d_in[0];   // [B,L,H]
  const float* in_up_w  = (const float*)d_in[1];   // [8*DI, H]
  const float* in_dn_w  = (const float*)d_in[2];   // [2*DI, 8*DI]
  const float* conv_w   = (const float*)d_in[3];   // [DI,1,K]
  const float* conv_b   = (const float*)d_in[4];   // [DI]
  const float* xproj_w  = (const float*)d_in[5];   // [R+2N, DI]
  const float* dt_w     = (const float*)d_in[6];   // [DI, R]
  const float* dt_b     = (const float*)d_in[7];   // [DI]
  const float* A_log    = (const float*)d_in[8];   // [DI, N]
  const float* Dv       = (const float*)d_in[9];   // [DI]
  const float* out_up_w = (const float*)d_in[10];  // [4*DI, DI]
  const float* out_dn_w = (const float*)d_in[11];  // [H, 4*DI]
  float* outp = (float*)d_out;                     // [B,L,H]

  const int N1 = 8 * DIc;       // 12288
  const int N2 = 2 * DIc;       // 3072
  const int N3 = Rc + 2 * Nst;  // 80
  const int N5 = 4 * DIc;       // 6144

  // bump allocator over d_ws
  size_t off = 0;
  auto alloc = [&](size_t bytes) -> void* {
    off = (off + 255) & ~(size_t)255;
    void* p = (char*)d_ws + off;
    off += bytes;
    return p;
  };

  bf16_t* x_bf    = (bf16_t*)alloc((size_t)BLc * Hc * 2);
  bf16_t* w1_bf   = (bf16_t*)alloc((size_t)N1 * Hc * 2);
  bf16_t* c1_bf   = (bf16_t*)alloc((size_t)BLc * N1 * 2);
  bf16_t* w2_bf   = (bf16_t*)alloc((size_t)N2 * N1 * 2);
  float*  proj_f  = (float*) alloc((size_t)BLc * N2 * 4);
  float*  hsc_f   = (float*) alloc((size_t)BLc * DIc * 4);
  bf16_t* hsc_bf  = (bf16_t*)alloc((size_t)BLc * DIc * 2);
  bf16_t* w3_bf   = (bf16_t*)alloc((size_t)N3 * DIc * 2);
  float*  ssmp_f  = (float*) alloc((size_t)BLc * N3 * 4);
  bf16_t* ts_bf   = (bf16_t*)alloc((size_t)BLc * KPAD * 2);
  bf16_t* dtw_bf  = (bf16_t*)alloc((size_t)DIc * KPAD * 2);
  float*  dt_f    = (float*) alloc((size_t)BLc * DIc * 4);
  bf16_t* y_bf    = (bf16_t*)alloc((size_t)BLc * DIc * 2);
  bf16_t* w5_bf   = (bf16_t*)alloc((size_t)N5 * DIc * 2);
  bf16_t* t1_bf   = (bf16_t*)alloc((size_t)BLc * N5 * 2);
  bf16_t* w6_bf   = (bf16_t*)alloc((size_t)Hc * N5 * 2);

  const int TB = 256;
  auto cvt = [&](const float* src, bf16_t* dst, int n) {
    f32_to_bf16_kernel<<<ceil_div(n, TB), TB, 0, stream>>>(src, dst, n);
  };

  // ---- precision casts ----
  cvt(x,        x_bf,  BLc * Hc);
  cvt(in_up_w,  w1_bf, N1 * Hc);
  cvt(in_dn_w,  w2_bf, N2 * N1);
  cvt(xproj_w,  w3_bf, N3 * DIc);
  cvt(out_up_w, w5_bf, N5 * DIc);
  cvt(out_dn_w, w6_bf, Hc * N5);
  pad_dtw_kernel<<<ceil_div(DIc * KPAD, TB), TB, 0, stream>>>(dt_w, dtw_bf);

  // ---- G1: silu(x @ in_up^T) -> bf16 ----
  gemm_bf16_wmma<ACT_SILU><<<dim3(ceil_div(N1, BN), BLc / BM), 256, 0, stream>>>(
      x_bf, w1_bf, nullptr, nullptr, c1_bf, BLc, N1, Hc);

  // ---- G2: proj = c1 @ in_down^T -> fp32 [BL, 2*DI] ----
  gemm_bf16_wmma<ACT_NONE><<<dim3(ceil_div(N2, BN), BLc / BM), 256, 0, stream>>>(
      c1_bf, w2_bf, nullptr, proj_f, nullptr, BLc, N2, N1);

  // ---- depthwise causal conv + SiLU ----
  conv_silu_kernel<<<ceil_div(Bc * Lc * DIc, TB), TB, 0, stream>>>(
      proj_f, conv_w, conv_b, hsc_f, hsc_bf);

  // ---- G3: ssm_p = hs @ x_proj^T -> fp32 [BL, 80] ----
  gemm_bf16_wmma<ACT_NONE><<<dim3(ceil_div(N3, BN), BLc / BM), 256, 0, stream>>>(
      hsc_bf, w3_bf, nullptr, ssmp_f, nullptr, BLc, N3, DIc);

  // ---- dt = softplus(ts @ dt_w^T + dt_b) (K padded 48->64) ----
  pad_ts_kernel<<<ceil_div(BLc * KPAD, TB), TB, 0, stream>>>(ssmp_f, ts_bf);
  gemm_bf16_wmma<ACT_SOFTPLUS><<<dim3(ceil_div(DIc, BN), BLc / BM), 256, 0, stream>>>(
      ts_bf, dtw_bf, dt_b, dt_f, nullptr, BLc, DIc, KPAD);

  // ---- selective scan + D residual + SiLU(gate) -> bf16 ----
  scan_kernel<<<dim3(DIc / 256, Bc), 256, 0, stream>>>(
      ssmp_f, dt_f, hsc_f, proj_f, A_log, Dv, y_bf);

  // ---- G5: silu(y @ out_up^T) -> bf16 ----
  gemm_bf16_wmma<ACT_SILU><<<dim3(ceil_div(N5, BN), BLc / BM), 256, 0, stream>>>(
      y_bf, w5_bf, nullptr, nullptr, t1_bf, BLc, N5, DIc);

  // ---- G6: out = t1 @ out_down^T -> fp32 d_out ----
  gemm_bf16_wmma<ACT_NONE><<<dim3(ceil_div(Hc, BN), BLc / BM), 256, 0, stream>>>(
      t1_bf, w6_bf, nullptr, outp, nullptr, BLc, Hc, N5);

  (void)in_sizes; (void)n_in; (void)out_size; (void)ws_size;
}
